// PerceiverAttention_2903397892389
// MI455X (gfx1250) — matrix operand order repaired and checked by reading, not compiled, not run on hardware
//
#include <hip/hip_runtime.h>

#define B_     8
#define N1_    4096
#define N2_    256
#define D_     1024
#define HEADS_ 16
#define DH_    64
#define LKV_   (N1_ + N2_)      /* 4352 */
#define INNER_ (HEADS_ * DH_)   /* 1024 */
#define ATTN_SCALE 0.35355339059327373f /* 64^-0.25, folded into q and k */

typedef __bf16 v8bf  __attribute__((ext_vector_type(8)));
typedef __bf16 v16bf __attribute__((ext_vector_type(16)));
typedef float  v8f   __attribute__((ext_vector_type(8)));

static __device__ __forceinline__ v8f vzero8() {
  v8f z;
#pragma unroll
  for (int i = 0; i < 8; ++i) z[i] = 0.0f;
  return z;
}

// WMMA 16x16x32 bf16, fp32 accumulate
static __device__ __forceinline__ v8f wmma_bf16(v16bf a, v16bf b, v8f c) {
  return __builtin_amdgcn_wmma_f32_16x16x32_bf16(false, a, false, b, (short)0, c,
                                                 false, false);
}

// Build the 16-bit A-matrix operand in contiguous K order (matches B):
// lanes 0-15 : elems 0-15 = K 0..15
// lanes 16-31: elems 0-15 = K 16..31
static __device__ __forceinline__ v16bf load_a16(const __bf16* base, int half) {
  const v8bf lo = *(const v8bf*)(base + 16 * half);
  const v8bf hi = *(const v8bf*)(base + 16 * half + 8);
  v16bf a;
#pragma unroll
  for (int e = 0; e < 8; ++e) { a[e] = lo[e]; a[8 + e] = hi[e]; }
  return a;
}

// ---------------------------------------------------------------------------
// fp32 -> bf16 weight conversion
// ---------------------------------------------------------------------------
__global__ void __launch_bounds__(256)
f2bf_kernel(const float* __restrict__ s, __bf16* __restrict__ d, int n) {
  int i = blockIdx.x * blockDim.x + threadIdx.x;
  if (i < n) d[i] = (__bf16)s[i];
}

// ---------------------------------------------------------------------------
// LayerNorm(x) -> bf16 rows [0,4096) of kvin[b, 0:4352, D]
// ---------------------------------------------------------------------------
__global__ void __launch_bounds__(256)
ln_x_kernel(const float* __restrict__ X, const float* __restrict__ w,
            const float* __restrict__ bia, __bf16* __restrict__ out) {
  __shared__ float ssum[8], ssq[8];
  __shared__ float smu, srv;
  const int row = blockIdx.x;            // 0..B*N1-1
  const int bb = row >> 12, l = row & (N1_ - 1);
  const float4* in = (const float4*)(X + (size_t)row * D_);
  __bf16* o = out + ((size_t)bb * LKV_ + l) * D_;
  const int t = threadIdx.x;
  float4 v = in[t];
  float s = v.x + v.y + v.z + v.w;
  float q = v.x * v.x + v.y * v.y + v.z * v.z + v.w * v.w;
#pragma unroll
  for (int m = 16; m >= 1; m >>= 1) {
    s += __shfl_xor(s, m, 32);
    q += __shfl_xor(q, m, 32);
  }
  if ((t & 31) == 0) { ssum[t >> 5] = s; ssq[t >> 5] = q; }
  __syncthreads();
  if (t == 0) {
    float S = 0.f, Q = 0.f;
    for (int i = 0; i < 8; ++i) { S += ssum[i]; Q += ssq[i]; }
    const float mu = S * (1.0f / D_);
    smu = mu;
    srv = rsqrtf(Q * (1.0f / D_) - mu * mu + 1e-5f);
  }
  __syncthreads();
  const float mu = smu, rv = srv;
  const int d = t * 4;
  o[d + 0] = (__bf16)((v.x - mu) * rv * w[d + 0] + bia[d + 0]);
  o[d + 1] = (__bf16)((v.y - mu) * rv * w[d + 1] + bia[d + 1]);
  o[d + 2] = (__bf16)((v.z - mu) * rv * w[d + 2] + bia[d + 2]);
  o[d + 3] = (__bf16)((v.w - mu) * rv * w[d + 3] + bia[d + 3]);
}

// ---------------------------------------------------------------------------
// LayerNorm(latents) * (1+scale) + shift -> bf16 rows [4096,4352) of kvin
// ---------------------------------------------------------------------------
__global__ void __launch_bounds__(256)
ln_lat_kernel(const float* __restrict__ X, const float* __restrict__ w,
              const float* __restrict__ bia, const float* __restrict__ scale,
              const float* __restrict__ shift, __bf16* __restrict__ out) {
  __shared__ float ssum[8], ssq[8];
  __shared__ float smu, srv;
  const int row = blockIdx.x;            // 0..B*N2-1
  const int bb = row >> 8, j = row & (N2_ - 1);
  const float4* in = (const float4*)(X + (size_t)row * D_);
  __bf16* o = out + ((size_t)bb * LKV_ + N1_ + j) * D_;
  const float* sc = scale + (size_t)bb * D_;
  const float* sh = shift + (size_t)bb * D_;
  const int t = threadIdx.x;
  float4 v = in[t];
  float s = v.x + v.y + v.z + v.w;
  float q = v.x * v.x + v.y * v.y + v.z * v.z + v.w * v.w;
#pragma unroll
  for (int m = 16; m >= 1; m >>= 1) {
    s += __shfl_xor(s, m, 32);
    q += __shfl_xor(q, m, 32);
  }
  if ((t & 31) == 0) { ssum[t >> 5] = s; ssq[t >> 5] = q; }
  __syncthreads();
  if (t == 0) {
    float S = 0.f, Q = 0.f;
    for (int i = 0; i < 8; ++i) { S += ssum[i]; Q += ssq[i]; }
    const float mu = S * (1.0f / D_);
    smu = mu;
    srv = rsqrtf(Q * (1.0f / D_) - mu * mu + 1e-5f);
  }
  __syncthreads();
  const float mu = smu, rv = srv;
  const int d = t * 4;
  float y0 = (v.x - mu) * rv * w[d + 0] + bia[d + 0];
  float y1 = (v.y - mu) * rv * w[d + 1] + bia[d + 1];
  float y2 = (v.z - mu) * rv * w[d + 2] + bia[d + 2];
  float y3 = (v.w - mu) * rv * w[d + 3] + bia[d + 3];
  o[d + 0] = (__bf16)(y0 * (1.0f + sc[d + 0]) + sh[d + 0]);
  o[d + 1] = (__bf16)(y1 * (1.0f + sc[d + 1]) + sh[d + 1]);
  o[d + 2] = (__bf16)(y2 * (1.0f + sc[d + 2]) + sh[d + 2]);
  o[d + 3] = (__bf16)(y3 * (1.0f + sc[d + 3]) + sh[d + 3]);
}

// ---------------------------------------------------------------------------
// Generic WMMA GEMM: C[m,n] = sum_k A[m,k] * W[n,k]
// block = 256 thr = 8 waves; wave computes 16(M) x 64(N); block tile 32x256.
// MODE 0 (KV): A=kvin rows direct; epilogue scatters K->[b,h,l,dh]*scale and
//              V->[b,h,dh,l] (transposed for the PV WMMA A-operand).
// MODE 1 (Q) : A rows are modulated latents inside kvin; out q[b,h,l,dh]*scale
// MODE 2 (O) : plain fp32 row-major store to d_out.
// ---------------------------------------------------------------------------
template <int MODE>
__global__ void __launch_bounds__(256)
gemm_wmma(const __bf16* __restrict__ A, const __bf16* __restrict__ W,
          __bf16* __restrict__ OB, float* __restrict__ OF, int K) {
  const int lane = threadIdx.x & 31;
  const int wave = threadIdx.x >> 5;
  const int half = lane >> 4;
  const int l16  = lane & 15;
  const int m0 = blockIdx.y * 32 + (wave & 1) * 16;
  const int n0 = blockIdx.x * 256 + (wave >> 1) * 64;

  const int mrow = m0 + l16;   // this lane's A row (A-operand layout)
  const __bf16* arow;
  if (MODE == 1) {
    const int bb = mrow >> 8, j = mrow & (N2_ - 1);
    arow = A + (size_t)(bb * LKV_ + N1_ + j) * D_;
  } else {
    arow = A + (size_t)mrow * K;
  }

  v8f acc[4];
#pragma unroll
  for (int j = 0; j < 4; ++j) acc[j] = vzero8();

  for (int k0 = 0; k0 < K; k0 += 32) {
    __builtin_prefetch(arow + k0 + 128, 0, 1);  // global_prefetch_b8
    const v16bf a = load_a16(arow + k0, half);
#pragma unroll
    for (int j = 0; j < 4; ++j) {
      // B-operand: lane = output column n, 16 contiguous K per half
      const v16bf bmat =
          *(const v16bf*)(W + (size_t)(n0 + 16 * j + l16) * K + k0 + 16 * half);
      acc[j] = wmma_bf16(a, bmat, acc[j]);
    }
  }

  // D layout: lanes 0-15 -> M = i, lanes 16-31 -> M = 8+i; column = lane&15
#pragma unroll
  for (int j = 0; j < 4; ++j) {
    const int n = n0 + 16 * j + l16;
#pragma unroll
    for (int i = 0; i < 8; ++i) {
      const int m = m0 + 8 * half + i;
      const float v = acc[j][i];
      if (MODE == 0) {
        const int bb = m / LKV_, l = m % LKV_;
        if (n < INNER_) {   // K half -> [b,h,l,dh], pre-scaled
          const int hh = n >> 6, dh = n & 63;
          OB[(((size_t)(bb * HEADS_ + hh) * LKV_ + l) << 6) + dh] =
              (__bf16)(v * ATTN_SCALE);
        } else {            // V half -> transposed [b,h,dh,l]
          const int e2 = n - INNER_;
          const int hh = e2 >> 6, dh = e2 & 63;
          OB[(size_t)B_ * HEADS_ * LKV_ * DH_ +
             ((size_t)(bb * HEADS_ + hh) * DH_ + dh) * LKV_ + l] = (__bf16)v;
        }
      } else if (MODE == 1) {
        const int bb = m >> 8, l = m & (N2_ - 1);
        const int hh = n >> 6, dh = n & 63;
        OB[(((size_t)(bb * HEADS_ + hh) * N2_ + l) << 6) + dh] =
            (__bf16)(v * ATTN_SCALE);
      } else {
        OF[(size_t)m * D_ + n] = v;
      }
    }
  }
}

// ---------------------------------------------------------------------------
// Flash attention, fully transposed so softmax rows live in lanes:
//   logits^T = K @ Q^T   (D column = query = lane; per-lane softmax state)
//   O^T      = V^T @ P^T (rescale & 1/sum are per-lane scalars)
// Key rows are permuted per WMMA tile (tile s row r -> key (r&7)+16*(r>>3)+8s)
// so each lane's exp() results land exactly in the B-operand layout of P^T:
// no LDS, no barriers, no cross-lane traffic except two shfl_xor(16) per block.
// ---------------------------------------------------------------------------
__global__ void __launch_bounds__(128)
attn_wmma(const __bf16* __restrict__ Q, const __bf16* __restrict__ Kh,
          const __bf16* __restrict__ Vt, __bf16* __restrict__ Ao) {
  const int lane = threadIdx.x & 31;
  const int wave = threadIdx.x >> 5;
  const int half = lane >> 4;
  const int l16  = lane & 15;
  const int bh = blockIdx.x >> 2;                    // 0..127
  const int qt = ((blockIdx.x & 3) << 2) + wave;     // 0..15
  const int b  = bh >> 4;
  const int h  = bh & (HEADS_ - 1);
  const int q0 = qt << 4;

  // Q^T B-operand (loop-invariant): lane = query column, 16 contiguous dh/half
  const __bf16* qcol = Q + (((size_t)bh * N2_ + q0 + l16) << 6);
  const v16bf bq0 = *(const v16bf*)(qcol + 16 * half);       // K = dh 0..31
  const v16bf bq1 = *(const v16bf*)(qcol + 32 + 16 * half);  // K = dh 32..63

  const __bf16* kbase = Kh + ((size_t)bh * LKV_ << 6);
  const __bf16* vbase = Vt + (size_t)bh * DH_ * LKV_;

  // permuted key row for this lane: tile s row l16 -> key kb + koff + 8*s
  const int koff0 = (l16 & 7) + 16 * (l16 >> 3);
  const int koff1 = koff0 + 8;

  float rm = -3.0e38f, rs = 0.0f;   // per-lane (= per-query) softmax state
  v8f o[4];
#pragma unroll
  for (int j = 0; j < 4; ++j) o[j] = vzero8();

  for (int kb = 0; kb < LKV_; kb += 32) {
    // logits^T tiles: c0 keys {kb+16*half+0..7}, c1 keys {kb+16*half+8..15}
    const __bf16* kr0 = kbase + (((size_t)(kb + koff0)) << 6);
    const __bf16* kr1 = kbase + (((size_t)(kb + koff1)) << 6);
    v8f c0 = vzero8(), c1 = vzero8();
    c0 = wmma_bf16(load_a16(kr0, half), bq0, c0);
    c0 = wmma_bf16(load_a16(kr0 + 32, half), bq1, c0);
    c1 = wmma_bf16(load_a16(kr1, half), bq0, c1);
    c1 = wmma_bf16(load_a16(kr1 + 32, half), bq1, c1);

    // per-lane row max over this lane's 16 keys, then cross-half combine
    float t = fmaxf(c0[0], c1[0]);
#pragma unroll
    for (int i = 1; i < 8; ++i) t = fmaxf(t, fmaxf(c0[i], c1[i]));
    t = fmaxf(t, __shfl_xor(t, 16, 32));
    const float nm = fmaxf(rm, t);
    const float f = __expf(rm - nm);
    rm = nm;

    float p0[8], p1[8], ss = 0.0f;
#pragma unroll
    for (int i = 0; i < 8; ++i) {
      p0[i] = __expf(c0[i] - nm);
      p1[i] = __expf(c1[i] - nm);
      ss += p0[i] + p1[i];
    }
    ss += __shfl_xor(ss, 16, 32);
    rs = rs * f + ss;
#pragma unroll
    for (int j = 0; j < 4; ++j)
#pragma unroll
      for (int i = 0; i < 8; ++i) o[j][i] = o[j][i] * f;

    // P^T B-operand: lane half hf holds keys K = 16*hf + {0..7 (c0), 8..15 (c1)}
    v16bf pb;
#pragma unroll
    for (int i = 0; i < 8; ++i) {
      pb[i]     = (__bf16)p0[i];
      pb[8 + i] = (__bf16)p1[i];
    }

    // O^T += V^T @ P^T : A-operand rows = dh (transposed V), K = 32 keys
#pragma unroll
    for (int j = 0; j < 4; ++j) {
      const v16bf av =
          load_a16(vbase + (size_t)(16 * j + l16) * LKV_ + kb, half);
      o[j] = wmma_bf16(av, pb, o[j]);
    }
  }

  // O^T D-layout: col = query = lane16, VGPR i -> dh = 16j + 8*half + i
  const float inv = 1.0f / rs;
#pragma unroll
  for (int j = 0; j < 4; ++j)
#pragma unroll
    for (int i = 0; i < 8; ++i)
      Ao[(size_t)(b * N2_ + q0 + l16) * INNER_ + h * DH_ + 16 * j + 8 * half +
         i] = (__bf16)(o[j][i] * inv);
}

// ---------------------------------------------------------------------------
extern "C" void kernel_launch(void* const* d_in, const int* in_sizes, int n_in,
                              void* d_out, int out_size, void* d_ws,
                              size_t ws_size, hipStream_t stream) {
  (void)in_sizes; (void)n_in; (void)out_size; (void)ws_size;
  const float* x     = (const float*)d_in[0];
  const float* lat   = (const float*)d_in[1];
  const float* shift = (const float*)d_in[2];
  const float* scale = (const float*)d_in[3];
  const float* ln1w  = (const float*)d_in[4];
  const float* ln1b  = (const float*)d_in[5];
  const float* ln2w  = (const float*)d_in[6];
  const float* ln2b  = (const float*)d_in[7];
  const float* Wq    = (const float*)d_in[8];
  const float* Wkv   = (const float*)d_in[9];
  const float* Wo    = (const float*)d_in[10];
  float* out = (float*)d_out;

  // Workspace layout (bf16 elements), ~230 MB total
  __bf16* ws   = (__bf16*)d_ws;
  __bf16* kvin = ws;                                            // 34816*1024
  __bf16* wkvb = kvin + (size_t)B_ * LKV_ * D_;                 // 2048*1024
  __bf16* wqb  = wkvb + (size_t)2 * INNER_ * D_;                // 1024*1024
  __bf16* wob  = wqb  + (size_t)INNER_ * D_;                    // 1024*1024
  __bf16* kbuf = wob  + (size_t)D_ * INNER_;                    // k + v (contig)
  __bf16* vtb  = kbuf + (size_t)B_ * HEADS_ * LKV_ * DH_;
  __bf16* qbuf = vtb  + (size_t)B_ * HEADS_ * LKV_ * DH_;
  __bf16* aout = qbuf + (size_t)B_ * HEADS_ * N2_ * DH_;        // 2048*1024

  // 1) weights fp32 -> bf16
  f2bf_kernel<<<(2 * INNER_ * D_ + 255) / 256, 256, 0, stream>>>(
      Wkv, wkvb, 2 * INNER_ * D_);
  f2bf_kernel<<<(INNER_ * D_ + 255) / 256, 256, 0, stream>>>(Wq, wqb,
                                                             INNER_ * D_);
  f2bf_kernel<<<(D_ * INNER_ + 255) / 256, 256, 0, stream>>>(Wo, wob,
                                                             D_ * INNER_);

  // 2) layernorms -> bf16 kvin
  ln_x_kernel<<<B_ * N1_, 256, 0, stream>>>(x, ln1w, ln1b, kvin);
  ln_lat_kernel<<<B_ * N2_, 256, 0, stream>>>(lat, ln2w, ln2b, scale, shift,
                                              kvin);

  // 3) KV projection: M=34816, N=2048, K=1024 -> k[b,h,l,dh]*s, vT[b,h,dh,l]
  gemm_wmma<0><<<dim3(2 * INNER_ / 256, B_ * LKV_ / 32), 256, 0, stream>>>(
      kvin, wkvb, kbuf, nullptr, D_);

  // 4) Q projection: M=2048, N=1024, K=1024 -> q[b,h,l,dh]*s
  gemm_wmma<1><<<dim3(INNER_ / 256, B_ * N2_ / 32), 256, 0, stream>>>(
      kvin, wqb, qbuf, nullptr, D_);

  // 5) flash attention (transposed, LDS-free) -> aout[b, n2, inner] (bf16)
  attn_wmma<<<B_ * HEADS_ * 4, 128, 0, stream>>>(qbuf, kbuf, vtb, aout);

  // 6) output projection: M=2048, N=1024, K=1024 -> fp32 d_out
  gemm_wmma<2><<<dim3(D_ / 256, B_ * N2_ / 32), 256, 0, stream>>>(
      aout, wob, nullptr, out, INNER_);
}
